// QuantLSTMModelStream_81381040325289
// MI455X (gfx1250) — 8-bit matrix operand order repaired and checked by reading, compiled, not run on hardware
//
#include <hip/hip_runtime.h>
#include <stdint.h>

#define HID   20
#define INP   10
#define G4    80
#define BATCH 1024
#define TLEN  1024
#define NBLK  16
#define ROWS_PER_BLK 64
#define NTHREADS 128

typedef __attribute__((ext_vector_type(8))) int          v8i;
typedef __attribute__((ext_vector_type(4))) int          i32x4_t;
typedef __attribute__((ext_vector_type(8))) int          i32x8_t;
typedef __attribute__((ext_vector_type(4))) unsigned int u32x4_t;

__device__ __forceinline__ unsigned umaxu(unsigned a, unsigned b) { return a > b ? a : b; }

__device__ __forceinline__ float po2_from_bits(unsigned bits) {
  float m = fmaxf(__uint_as_float(bits), 1e-8f);
  return exp2f(ceilf(log2f(m))) * 0.0078125f;
}

__device__ __forceinline__ float fqv(float x, float s) {
  float r = rintf(x / s);
  r = fminf(fmaxf(r, -128.0f), 127.0f);
  return r * s;
}

__device__ __forceinline__ float sigmoidf_(float v) { return 1.0f / (1.0f + __expf(-v)); }

__device__ __forceinline__ unsigned load_u_agent(unsigned* p) {
  return __hip_atomic_load(p, __ATOMIC_RELAXED, __HIP_MEMORY_SCOPE_AGENT);
}

__device__ __forceinline__ unsigned block_reduce_max(unsigned v, unsigned* sred, int tid) {
#pragma unroll
  for (int off = 16; off > 0; off >>= 1)
    v = umaxu(v, (unsigned)__shfl_xor((int)v, off, 32));
  __syncthreads();
  if ((tid & 31) == 0) sred[tid >> 5] = v;
  __syncthreads();
  return umaxu(umaxu(sred[0], sred[1]), umaxu(sred[2], sred[3]));
}

__device__ __forceinline__ void grid_barrier(unsigned* cnt, unsigned* gen, int tid) {
  __syncthreads();
  if (tid == 0) {
    __threadfence();
    unsigned g = __hip_atomic_load(gen, __ATOMIC_ACQUIRE, __HIP_MEMORY_SCOPE_AGENT);
    unsigned a = __hip_atomic_fetch_add(cnt, 1u, __ATOMIC_ACQ_REL, __HIP_MEMORY_SCOPE_AGENT);
    if (a == (unsigned)(NBLK - 1)) {
      __hip_atomic_store(cnt, 0u, __ATOMIC_RELAXED, __HIP_MEMORY_SCOPE_AGENT);
      __hip_atomic_fetch_add(gen, 1u, __ATOMIC_RELEASE, __HIP_MEMORY_SCOPE_AGENT);
    } else {
      while (__hip_atomic_load(gen, __ATOMIC_ACQUIRE, __HIP_MEMORY_SCOPE_AGENT) == g)
        __builtin_amdgcn_s_sleep(2);
    }
  }
  __syncthreads();
}

__device__ __forceinline__ void tdm_load_xtile(const float* xg, unsigned lds_off,
                                               int b0, int t) {
  unsigned long long ga = (unsigned long long)(uintptr_t)xg +
      (unsigned long long)((unsigned)b0 * (unsigned)TLEN + (unsigned)t) * (INP * 4ull);
  u32x4_t g0; i32x8_t g1; i32x4_t g2, g3; i32x8_t g4;
  g0[0] = 1u;
  g0[1] = lds_off;
  g0[2] = (unsigned)ga;
  g0[3] = (unsigned)((ga >> 32) & 0x01FFFFFFull) | (2u << 30);
  g1[0] = (int)(2u << 16);
  g1[1] = (int)((unsigned)INP << 16);
  g1[2] = (int)((unsigned)ROWS_PER_BLK << 16);
  g1[3] = (int)((unsigned)INP << 16);
  g1[4] = (int)(unsigned)ROWS_PER_BLK;
  g1[5] = (int)(unsigned)(TLEN * INP);
  g1[6] = 0;
  g1[7] = (int)(unsigned)(((unsigned long long)TLEN * INP * ROWS_PER_BLK) >> 16);
  g2[0] = 1; g2[1] = 1; g2[2] = 0; g2[3] = 0;
  g3[0] = 0; g3[1] = (int)(1u << 16); g3[2] = 0; g3[3] = 0;
#pragma unroll
  for (int i = 0; i < 8; ++i) g4[i] = 0;
  __builtin_amdgcn_tensor_load_to_lds(g0, g1, g2, g3, g4, 0);
}

__global__ void ws_init_kernel(unsigned* p, int n) {
  int i = blockIdx.x * blockDim.x + threadIdx.x;
  if (i < n) p[i] = 0u;
}

__global__ __launch_bounds__(NTHREADS, 1)
void qlstm_kernel(const float* __restrict__ x, const float* __restrict__ Wih,
                  const float* __restrict__ Whh, float* __restrict__ out,
                  unsigned* __restrict__ gbar) {
  __shared__ float       xle[2][ROWS_PER_BLK * INP];
  __shared__ float       glds[ROWS_PER_BLK * G4];
  __shared__ float       wifq[G4 * INP];
  __shared__ signed char whq8[G4 * HID];
  __shared__ signed char hq8[ROWS_PER_BLK * 64];
  __shared__ unsigned    sred[4];

  const int tid  = threadIdx.x;
  const int lane = tid & 31;
  const int w16  = (tid >> 5) * 16;
  const int b0   = blockIdx.x * ROWS_PER_BLK;
  unsigned* cnt   = gbar + 0;
  unsigned* gen   = gbar + 1;
  unsigned* slots = gbar + 2;

  float lm = 0.0f;
  for (int i = tid; i < G4 * INP; i += NTHREADS) lm = fmaxf(lm, fabsf(Wih[i]));
  for (int i = tid; i < G4 * HID; i += NTHREADS) lm = fmaxf(lm, fabsf(Whh[i]));
  const unsigned wb = block_reduce_max(__float_as_uint(lm), sred, tid);
  const float wscale = po2_from_bits(wb);
  const float inv_ws = 1.0f / wscale;
  for (int i = tid; i < G4 * INP; i += NTHREADS) {
    float q = fminf(fmaxf(rintf(Wih[i] * inv_ws), -128.0f), 127.0f);
    wifq[i] = q * wscale;
  }
  for (int i = tid; i < G4 * HID; i += NTHREADS) {
    float q = fminf(fmaxf(rintf(Whh[i] * inv_ws), -128.0f), 127.0f);
    whq8[i] = (signed char)(int)q;
  }
  for (int i = tid; i < (ROWS_PER_BLK * 64) / 4; i += NTHREADS) ((int*)hq8)[i] = 0;
  __syncthreads();

  const int col = lane & 15;
  float wiR[5][INP];
#pragma unroll
  for (int n = 0; n < 5; ++n)
#pragma unroll
    for (int i = 0; i < INP; ++i) wiR[n][i] = wifq[(n * 16 + col) * INP + i];

  v8i whB[5];
#pragma unroll
  for (int n = 0; n < 5; ++n) {
    const signed char* wr = &whq8[(n * 16 + col) * HID];
    const int kb = (lane & 16) ? 8 : 0;
#pragma unroll
    for (int v = 0; v < 4; ++v) {
      unsigned pk = 0;
#pragma unroll
      for (int b = 0; b < 4; ++b) {
        int K = (v < 2) ? (kb + v * 4 + b) : (16 + kb + (v - 2) * 4 + b);
        int q = (K < HID) ? (int)wr[K] : 0;
        pk |= ((unsigned)(q & 255)) << (8 * b);
      }
      whB[n][v] = (int)pk;
    }
#pragma unroll
    for (int v = 4; v < 8; ++v) whB[n][v] = 0;
  }

  float creg[10];
#pragma unroll
  for (int q = 0; q < 10; ++q) creg[q] = 0.0f;
  float hs_prev = 1.0f;

  if (tid < 32) tdm_load_xtile(x, (unsigned)(uintptr_t)&xle[0][0], b0, 0);

  const int erow  = lane >> 1;
  const int j0    = (lane & 1) * 10;
  const int gbase = (w16 + erow) * G4;

  for (int t = 0; t < TLEN; ++t) {
    if (tid < 32) __builtin_amdgcn_s_wait_tensorcnt(0);
    __syncthreads();
    if (tid < 32 && (t + 1) < TLEN)
      tdm_load_xtile(x, (unsigned)(uintptr_t)&xle[(t + 1) & 1][0], b0, t + 1);

    v8i A;
    {
      const int* hp = (const int*)&hq8[(w16 + (lane & 15)) * 64];
      const int lo = (lane & 16) ? 2 : 0;
      A[0] = hp[lo]; A[1] = hp[lo + 1]; A[2] = hp[4 + lo];
      A[3] = 0; A[4] = 0; A[5] = 0; A[6] = 0; A[7] = 0;
    }
    const float kscale = wscale * hs_prev;
    v8i cz;
#pragma unroll
    for (int v = 0; v < 8; ++v) cz[v] = 0;

    float gf[5][8];
#pragma unroll
    for (int n = 0; n < 5; ++n) {
      v8i s32 = __builtin_amdgcn_wmma_i32_16x16x64_iu8(true, A, true, whB[n], cz, false, false);
#pragma unroll
      for (int v = 0; v < 8; ++v) gf[n][v] = kscale * (float)s32[v];
    }

    const float* xb = &xle[t & 1][0];
    const int half = (lane & 16) ? 8 : 0;
#pragma unroll
    for (int v = 0; v < 8; ++v) {
      const float* xr = &xb[(w16 + v + half) * INP];
      float xv[INP];
#pragma unroll
      for (int i = 0; i < INP; ++i) xv[i] = xr[i];
#pragma unroll
      for (int n = 0; n < 5; ++n) {
        float acc = gf[n][v];
#pragma unroll
        for (int i = 0; i < INP; ++i) acc = fmaf(xv[i], wiR[n][i], acc);
        gf[n][v] = acc;
      }
    }

    float mg = 0.0f;
#pragma unroll
    for (int n = 0; n < 5; ++n)
#pragma unroll
      for (int v = 0; v < 8; ++v) {
        glds[(w16 + v + half) * G4 + n * 16 + col] = gf[n][v];
        mg = fmaxf(mg, fabsf(gf[n][v]));
      }

    unsigned* s0 = slots + (size_t)t * 8;
    unsigned bm = block_reduce_max(__float_as_uint(mg), sred, tid);
    if (tid == 0) atomicMax(s0, bm);
    grid_barrier(cnt, gen, tid);
    const float sgate = po2_from_bits(load_u_agent(s0));

    float mi = 0.f, mf = 0.f, mgt = 0.f, mo = 0.f;
#pragma unroll
    for (int q = 0; q < 10; ++q) {
      int j = j0 + q;
      float si = sigmoidf_(fqv(glds[gbase + j],        sgate));
      float sf = sigmoidf_(fqv(glds[gbase + 20 + j],   sgate));
      float sg = tanhf    (fqv(glds[gbase + 40 + j],   sgate));
      float so = sigmoidf_(fqv(glds[gbase + 60 + j],   sgate));
      glds[gbase + j] = si; glds[gbase + 20 + j] = sf;
      glds[gbase + 40 + j] = sg; glds[gbase + 60 + j] = so;
      mi = fmaxf(mi, fabsf(si)); mf = fmaxf(mf, fabsf(sf));
      mgt = fmaxf(mgt, fabsf(sg)); mo = fmaxf(mo, fabsf(so));
    }
    unsigned bi = block_reduce_max(__float_as_uint(mi),  sred, tid);
    unsigned bf = block_reduce_max(__float_as_uint(mf),  sred, tid);
    unsigned bg = block_reduce_max(__float_as_uint(mgt), sred, tid);
    unsigned bo = block_reduce_max(__float_as_uint(mo),  sred, tid);
    if (tid == 0) { atomicMax(s0 + 1, bi); atomicMax(s0 + 2, bf);
                    atomicMax(s0 + 3, bg); atomicMax(s0 + 4, bo); }
    grid_barrier(cnt, gen, tid);
    const float s_i = po2_from_bits(load_u_agent(s0 + 1));
    const float s_f = po2_from_bits(load_u_agent(s0 + 2));
    const float s_g = po2_from_bits(load_u_agent(s0 + 3));
    const float s_o = po2_from_bits(load_u_agent(s0 + 4));

    float mt = 0.0f;
#pragma unroll
    for (int q = 0; q < 10; ++q) {
      int j = j0 + q;
      float i_ = fqv(glds[gbase + j],        s_i);
      float f_ = fqv(glds[gbase + 20 + j],   s_f);
      float g_ = fqv(glds[gbase + 40 + j],   s_g);
      float o_ = fqv(glds[gbase + 60 + j],   s_o);
      float c  = fmaf(f_, creg[q], i_ * g_);
      creg[q] = c;
      float th = tanhf(c);
      glds[gbase + j]      = th;
      glds[gbase + 60 + j] = o_;
      mt = fmaxf(mt, fabsf(th));
    }
    unsigned bt = block_reduce_max(__float_as_uint(mt), sred, tid);
    if (tid == 0) atomicMax(s0 + 5, bt);
    grid_barrier(cnt, gen, tid);
    const float s_t = po2_from_bits(load_u_agent(s0 + 5));

    float ph[10]; float mh = 0.0f;
#pragma unroll
    for (int q = 0; q < 10; ++q) {
      int j = j0 + q;
      float p = glds[gbase + 60 + j] * fqv(glds[gbase + j], s_t);
      ph[q] = p; mh = fmaxf(mh, fabsf(p));
    }
    unsigned bh = block_reduce_max(__float_as_uint(mh), sred, tid);
    if (tid == 0) atomicMax(s0 + 6, bh);
    grid_barrier(cnt, gen, tid);
    const float s_h = po2_from_bits(load_u_agent(s0 + 6));
    const float inv_sh = 1.0f / s_h;

    const size_t obase = (size_t)(b0 + w16 + erow) * (size_t)(TLEN * HID)
                       + (size_t)t * HID + (size_t)j0;
#pragma unroll
    for (int q = 0; q < 10; ++q) {
      float r = fminf(fmaxf(rintf(ph[q] * inv_sh), -128.0f), 127.0f);
      hq8[(w16 + erow) * 64 + j0 + q] = (signed char)(int)r;
      out[obase + q] = r * s_h;
    }
    hs_prev = s_h;
  }
}

extern "C" void kernel_launch(void* const* d_in, const int* in_sizes, int n_in,
                              void* d_out, int out_size, void* d_ws, size_t ws_size,
                              hipStream_t stream) {
  (void)in_sizes; (void)n_in; (void)out_size; (void)ws_size;
  const float* x   = (const float*)d_in[0];
  const float* Wih = (const float*)d_in[1];
  const float* Whh = (const float*)d_in[2];
  float* out = (float*)d_out;
  unsigned* gbar = (unsigned*)d_ws;

  const int nz = 2 + 8 * TLEN;
  ws_init_kernel<<<(nz + 255) / 256, 256, 0, stream>>>(gbar, nz);
  qlstm_kernel<<<NBLK, NTHREADS, 0, stream>>>(x, Wih, Whh, out, gbar);
}
